// MultiHeadAttention_12335146074164
// MI455X (gfx1250) — matrix operand order repaired and checked by reading, not compiled, not run on hardware
//
#include <hip/hip_runtime.h>

typedef __bf16 bf16;
typedef __attribute__((ext_vector_type(8)))  __bf16 v8bf;
typedef __attribute__((ext_vector_type(16))) __bf16 v16bf;
typedef __attribute__((ext_vector_type(8)))  float  v8f;
typedef __attribute__((ext_vector_type(4)))  unsigned u32x4;
typedef __attribute__((ext_vector_type(4)))  int      i32x4;
typedef __attribute__((ext_vector_type(8)))  int      i32x8;

#define WMMA_BF16(A,B,C) \
  __builtin_amdgcn_wmma_f32_16x16x32_bf16(false,(A),false,(B),(short)0,(C),false,false)

#if defined(__gfx1250__) && __has_builtin(__builtin_amdgcn_tensor_load_to_lds)
#define HAVE_TDM 1
#if __has_include(<hip/amd_detail/amd_gfx1250_TDM.h>)
#define TDM_SIX_ARGS 1   // therock-10.0 headers -> 6-arg builtin
#else
#define TDM_SIX_ARGS 0   // ROCm 7.2 -> 5-arg builtin
#endif
#else
#define HAVE_TDM 0
#endif

constexpr int BATCH = 4;
constexpr int SEQ   = 2048;
constexpr int EMB   = 1024;
constexpr int NH    = 16;
constexpr int HDIM  = 64;
constexpr int MTOT  = BATCH * SEQ;   // 8192

static __device__ __forceinline__ bf16 f2bf(float f) {
  unsigned u = __builtin_bit_cast(unsigned, f);
  unsigned rnd = u + 0x7fffu + ((u >> 16) & 1u);
  unsigned short h = (unsigned short)(rnd >> 16);
  return __builtin_bit_cast(bf16, h);
}
static __device__ __forceinline__ float bf2f(bf16 b) {
  unsigned short h = __builtin_bit_cast(unsigned short, b);
  unsigned u = ((unsigned)h) << 16;
  return __builtin_bit_cast(float, u);
}
static __device__ __forceinline__ v16bf cat16(v8bf lo, v8bf hi) {
  v16bf o;
#pragma unroll
  for (int e = 0; e < 8; ++e) { o[e] = lo[e]; o[e + 8] = hi[e]; }
  return o;
}

// ---- Tensor Data Mover: async 2D tile (rows x cols, 2-byte elems) global -> LDS ----
// Descriptor packing per CDNA5 ISA 8.3/8.4: group0 = {count, lds_addr, global_addr, type=2},
// group1 = {data_size=2B, tensor_dim0/1, tile_dim0/1, tensor_dim0_stride}.
static __device__ __forceinline__ void tdm_load_2d(
    unsigned lds_off, const void* gptr,
    unsigned dim0, unsigned dim1, unsigned stride0,
    unsigned tile0, unsigned tile1) {
#if HAVE_TDM
  unsigned long long ga = (unsigned long long)gptr;
  u32x4 g0;
  g0[0] = 1u;                                  // count=1 (valid user descriptor)
  g0[1] = lds_off;                             // lds_addr (bytes)
  g0[2] = (unsigned)ga;                        // global_addr[31:0]
  g0[3] = (unsigned)(ga >> 32) | (2u << 30);   // global_addr[56:32] | type=2 ("image")

  i32x8 g1;
  g1[0] = 0x00010000;                          // data_size=1 -> 2-byte elements
  g1[1] = (int)((dim0 & 0xFFFFu) << 16);       // tensor_dim0[15:0] @ bits 63:48
  g1[2] = (int)((dim0 >> 16) | ((dim1 & 0xFFFFu) << 16));   // dim0[31:16] | dim1[15:0]
  g1[3] = (int)((dim1 >> 16) | (tile0 << 16)); // dim1[31:16] | tile_dim0
  g1[4] = (int)tile1;                          // tile_dim1 (tile_dim2 = 0)
  g1[5] = (int)stride0;                        // tensor_dim0_stride[31:0]
  g1[6] = 0;
  g1[7] = 0;

  i32x4 z4 = {0, 0, 0, 0};
#if TDM_SIX_ARGS
  i32x8 z8 = {0, 0, 0, 0, 0, 0, 0, 0};
  __builtin_amdgcn_tensor_load_to_lds(g0, g1, z4, z4, z8, 0);
#else
  __builtin_amdgcn_tensor_load_to_lds(g0, g1, z4, z4, 0);
#endif
#endif
}

// ---------------- fp32 -> bf16 conversion ----------------
__global__ void cvt_f32_bf16(const float* __restrict__ in, bf16* __restrict__ out, int n) {
  int i = (blockIdx.x * blockDim.x + threadIdx.x) * 4;
  if (i < n) {
    float4 v = *(const float4*)(in + i);
    out[i + 0] = f2bf(v.x); out[i + 1] = f2bf(v.y);
    out[i + 2] = f2bf(v.z); out[i + 3] = f2bf(v.w);
  }
}

// ---- fp32 [K,N] -> bf16 transposed [N,K] (weights, so GEMM B-tiles are plain row fetches) ----
__global__ void cvt_transpose_bf16(const float* __restrict__ in, bf16* __restrict__ out) {
  int idx = blockIdx.x * blockDim.x + threadIdx.x;  // EMB*EMB
  int n = idx & (EMB - 1);
  int k = idx >> 10;
  out[(size_t)n * EMB + k] = f2bf(in[(size_t)k * EMB + n]);
}

// ---------------- bf16 WMMA GEMM: C[M,N] = A[M,K] @ Bt[N,K]^T ----------------
// 256 threads (8 waves), 64x64 tile, K-step 32, TDM double-buffered tile staging.
template <bool BF16OUT>
__global__ __launch_bounds__(256) void gemm_bt_wmma(
    const bf16* __restrict__ A, const bf16* __restrict__ Bt,
    void* __restrict__ Cout, int M, int N, int K) {
  __shared__ __align__(32) bf16 As[2][64][32];   // [buf][m][k]
  __shared__ __align__(32) bf16 Bs[2][64][32];   // [buf][n][k]

  const int tid  = threadIdx.x;
  const int wave = tid >> 5, lane = tid & 31;
  const int hl   = lane >> 4, r = lane & 15;
  const int m0   = blockIdx.y * 64, n0 = blockIdx.x * 64;
  const int wm   = (wave & 3) * 16;      // wave row tile: 0/16/32/48
  const int wn   = (wave >> 2) * 32;     // wave col half: 0/32

  const int nkt = K >> 5;
  v8f acc0 = {}; v8f acc1 = {};

#if HAVE_TDM
  auto issue = [&](int t) {
    int k0 = t * 32, buf = t & 1;
    tdm_load_2d((unsigned)(size_t)&As[buf][0][0],
                A + (size_t)m0 * K + k0, (unsigned)K, (unsigned)M, (unsigned)K, 32, 64);
    tdm_load_2d((unsigned)(size_t)&Bs[buf][0][0],
                Bt + (size_t)n0 * K + k0, (unsigned)K, (unsigned)N, (unsigned)K, 32, 64);
  };
  if (wave == 0) issue(0);
#else
  const int idx = tid * 8;
  const int ar = idx >> 5, ak = idx & 31;   // staging coords (64x32 tile, 16B/thread)
#endif

  for (int t = 0; t < nkt; ++t) {
    const int buf = t & 1;
#if HAVE_TDM
    if (wave == 0) {
      if (t + 1 < nkt) {
        issue(t + 1);                                // prefetch next tile (other buffer)
        __builtin_amdgcn_s_wait_tensorcnt(2);        // tile t complete, t+1 in flight
      } else {
        __builtin_amdgcn_s_wait_tensorcnt(0);
      }
    }
    __syncthreads();
#else
    {
      int k0 = t * 32;
      *(v8bf*)&As[buf][ar][ak] = *(const v8bf*)(A  + (size_t)(m0 + ar) * K + k0 + ak);
      *(v8bf*)&Bs[buf][ar][ak] = *(const v8bf*)(Bt + (size_t)(n0 + ar) * K + k0 + ak);
    }
    __syncthreads();
#endif

    // A-frag: row = r; elem e -> k = hl*16+e
    v16bf af = cat16(*(const v8bf*)&As[buf][wm + r][hl * 16],
                     *(const v8bf*)&As[buf][wm + r][hl * 16 + 8]);
    // B-frag: col = r; elem e -> k = hl*16+e (contiguous 32B per lane)
    v16bf b0 = *(const v16bf*)&Bs[buf][wn + r][hl * 16];
    v16bf b1 = *(const v16bf*)&Bs[buf][wn + 16 + r][hl * 16];
    acc0 = WMMA_BF16(af, b0, acc0);
    acc1 = WMMA_BF16(af, b1, acc1);
    __syncthreads();   // all waves done with buf before TDM refills it next round
  }

#pragma unroll
  for (int i = 0; i < 8; ++i) {
    size_t row = (size_t)(m0 + wm + i + 8 * hl);
    size_t c = row * (size_t)N + n0 + wn + r;
    if constexpr (BF16OUT) {
      ((bf16*)Cout)[c]      = f2bf(acc0[i]);
      ((bf16*)Cout)[c + 16] = f2bf(acc1[i]);
    } else {
      ((float*)Cout)[c]      = acc0[i];
      ((float*)Cout)[c + 16] = acc1[i];
    }
  }
}

// ---------------- RoPE + head-split reshape for Q and K ----------------
__global__ void rope_reshape_qk(const bf16* __restrict__ qraw, const bf16* __restrict__ kraw,
                                bf16* __restrict__ Qr, bf16* __restrict__ Kr) {
  int idx = blockIdx.x * blockDim.x + threadIdx.x;  // B*H*S*32 = 2^22 threads
  int j = idx & 31;
  int s = (idx >> 5) & (SEQ - 1);
  int h = (idx >> 16) & (NH - 1);
  int b = idx >> 20;
  float freq = __expf(-(float)(2 * j) * (9.210340371976184f / 64.0f));  // ln(1e4)/dim
  float ang = (float)s * freq;
  float sn, cs;
  __sincosf(ang, &sn, &cs);
  size_t src = ((size_t)(b * SEQ + s)) * EMB + h * HDIM;
  size_t dst = (((size_t)(b * NH + h)) * SEQ + s) * HDIM;
  float q1 = bf2f(qraw[src + j]), q2 = bf2f(qraw[src + 32 + j]);
  Qr[dst + j]      = f2bf(cs * q1 - sn * q2);
  Qr[dst + 32 + j] = f2bf(sn * q1 + cs * q2);
  float k1 = bf2f(kraw[src + j]), k2 = bf2f(kraw[src + 32 + j]);
  Kr[dst + j]      = f2bf(cs * k1 - sn * k2);
  Kr[dst + 32 + j] = f2bf(sn * k1 + cs * k2);
}

// ---------------- V: [B*S, E] -> Vt [B,H,D,S] (transposed for P@V B-frags) ----------------
__global__ void transpose_v(const bf16* __restrict__ vraw, bf16* __restrict__ Vt) {
  int idx = blockIdx.x * blockDim.x + threadIdx.x;  // B*H*S*D = 2^23 threads
  int d = idx & (HDIM - 1);
  int s = (idx >> 6) & (SEQ - 1);
  int h = (idx >> 17) & (NH - 1);
  int b = idx >> 21;
  Vt[(((size_t)(b * NH + h)) * HDIM + d) * SEQ + s] =
      vraw[((size_t)(b * SEQ + s)) * EMB + h * HDIM + d];
}

// ---------------- causal flash attention (one wave per 16 query rows) ----------------
__global__ __launch_bounds__(128) void flash_attn(
    const bf16* __restrict__ Q, const bf16* __restrict__ Kr,
    const bf16* __restrict__ Vt, bf16* __restrict__ AO) {
  __shared__ __align__(32) bf16 Pt[4][16][32];  // per-wave prob tile [row][key]

  const int wave = threadIdx.x >> 5, lane = threadIdx.x & 31;
  const int hl = lane >> 4, r = lane & 15;
  const int bh = blockIdx.y;
  const int b = bh >> 4, h = bh & 15;
  const int q0 = (blockIdx.x * 4 + wave) * 16;

  const bf16* Qbh = Q  + (size_t)bh * SEQ * HDIM;
  const bf16* Kbh = Kr + (size_t)bh * SEQ * HDIM;
  const bf16* Vbh = Vt + (size_t)bh * HDIM * SEQ;

  // Q A-fragments, kept in registers for the whole row-tile (d 0..31 and 32..63)
  const bf16* qrow = Qbh + (size_t)(q0 + r) * HDIM;
  v16bf aq0 = cat16(*(const v8bf*)(qrow + hl * 16),      *(const v8bf*)(qrow + 8 + hl * 16));
  v16bf aq1 = cat16(*(const v8bf*)(qrow + 32 + hl * 16), *(const v8bf*)(qrow + 40 + hl * 16));

  float m[8], l[8];
  v8f o0 = {}, o1 = {}, o2 = {}, o3 = {};
#pragma unroll
  for (int i = 0; i < 8; ++i) { m[i] = -1e30f; l[i] = 0.f; }

  const int kend = q0 + 16;  // causal: keys 0..q0+15
  for (int kb = 0; kb < kend; kb += 32) {
    if (kb + 32 < kend) {
      __builtin_prefetch(Kbh + (size_t)(kb + 32 + r) * HDIM, 0, 1);
      __builtin_prefetch(Vbh + (size_t)r * SEQ + kb + 32, 0, 1);
    }
    // scores S = (Q K^T) for two 16-key groups; K_gemm = D = 64 -> 2 WMMAs each
    v8f s0 = {}, s1 = {};
    {
      const bf16* kr0 = Kbh + (size_t)(kb + r) * HDIM;
      s0 = WMMA_BF16(aq0, *(const v16bf*)(kr0 + hl * 16), s0);
      s0 = WMMA_BF16(aq1, *(const v16bf*)(kr0 + 32 + hl * 16), s0);
      const bf16* kr1 = Kbh + (size_t)(kb + 16 + r) * HDIM;
      s1 = WMMA_BF16(aq0, *(const v16bf*)(kr1 + hl * 16), s1);
      s1 = WMMA_BF16(aq1, *(const v16bf*)(kr1 + 32 + hl * 16), s1);
    }

    // online softmax per row (row = q0 + i + 8*hl, key col = lane r)
#pragma unroll
    for (int i = 0; i < 8; ++i) {
      int row = q0 + i + 8 * hl;
      float v0 = s0[i] * 0.125f;             // 1/sqrt(64)
      float v1 = s1[i] * 0.125f;
      if (kb + r > row)      v0 = -1e30f;    // causal mask
      if (kb + 16 + r > row) v1 = -1e30f;
      float mx = fmaxf(v0, v1);
#pragma unroll
      for (int t = 1; t < 16; t <<= 1) mx = fmaxf(mx, __shfl_xor(mx, t, 32));
      float mnew  = fmaxf(m[i], mx);
      float alpha = __expf(m[i] - mnew);
      float p0 = __expf(v0 - mnew), p1 = __expf(v1 - mnew);
      float rs = p0 + p1;
#pragma unroll
      for (int t = 1; t < 16; t <<= 1) rs += __shfl_xor(rs, t, 32);
      l[i] = l[i] * alpha + rs;
      m[i] = mnew;
      o0[i] *= alpha; o1[i] *= alpha; o2[i] *= alpha; o3[i] *= alpha;
      Pt[wave][i + 8 * hl][r]      = f2bf(p0);
      Pt[wave][i + 8 * hl][16 + r] = f2bf(p1);
    }

    // P (16x32) A-fragment from LDS; V B-frags straight from Vt (contiguous 32B/lane)
    v16bf pA = cat16(*(const v8bf*)&Pt[wave][r][hl * 16],
                     *(const v8bf*)&Pt[wave][r][hl * 16 + 8]);
    const bf16* vb = Vbh + (size_t)r * SEQ + kb + hl * 16;
    o0 = WMMA_BF16(pA, *(const v16bf*)(vb),            o0);
    o1 = WMMA_BF16(pA, *(const v16bf*)(vb + 16 * SEQ), o1);
    o2 = WMMA_BF16(pA, *(const v16bf*)(vb + 32 * SEQ), o2);
    o3 = WMMA_BF16(pA, *(const v16bf*)(vb + 48 * SEQ), o3);
  }

  // epilogue: normalize and store merged back to [B,S,E] bf16 for the out-proj GEMM
#pragma unroll
  for (int i = 0; i < 8; ++i) {
    float inv = 1.0f / l[i];
    size_t base = ((size_t)(b * SEQ + q0 + i + 8 * hl)) * EMB + h * HDIM + r;
    AO[base]      = f2bf(o0[i] * inv);
    AO[base + 16] = f2bf(o1[i] * inv);
    AO[base + 32] = f2bf(o2[i] * inv);
    AO[base + 48] = f2bf(o3[i] * inv);
  }
}

// ---------------- host launcher ----------------
extern "C" void kernel_launch(void* const* d_in, const int* in_sizes, int n_in,
                              void* d_out, int out_size, void* d_ws, size_t ws_size,
                              hipStream_t stream) {
  const float* x  = (const float*)d_in[0];
  const float* Wq = (const float*)d_in[1];
  const float* Wk = (const float*)d_in[2];
  const float* Wv = (const float*)d_in[3];
  const float* Wo = (const float*)d_in[4];

  char* p = (char*)d_ws;
  auto take = [&](size_t bytes) -> bf16* {
    bf16* q = (bf16*)p;
    p += (bytes + 255) & ~(size_t)255;
    return q;
  };
  const size_t XE = (size_t)MTOT * EMB;  // 8388608 elems
  const size_t WE = (size_t)EMB * EMB;   // 1048576 elems

  bf16* xb   = take(XE * 2);
  bf16* Wqt  = take(WE * 2);   // transposed bf16 weights [N,K]
  bf16* Wkt  = take(WE * 2);
  bf16* Wvt  = take(WE * 2);
  bf16* Wot  = take(WE * 2);
  bf16* qraw = take(XE * 2);
  bf16* kraw = take(XE * 2);
  bf16* vraw = take(XE * 2);
  bf16* Qr   = take(XE * 2);
  bf16* Kr   = take(XE * 2);
  bf16* Vt   = take(XE * 2);
  bf16* AO   = take(XE * 2);

  cvt_f32_bf16<<<(int)(XE / 1024), 256, 0, stream>>>(x, xb, (int)XE);
  cvt_transpose_bf16<<<(int)(WE / 256), 256, 0, stream>>>(Wq, Wqt);
  cvt_transpose_bf16<<<(int)(WE / 256), 256, 0, stream>>>(Wk, Wkt);
  cvt_transpose_bf16<<<(int)(WE / 256), 256, 0, stream>>>(Wv, Wvt);
  cvt_transpose_bf16<<<(int)(WE / 256), 256, 0, stream>>>(Wo, Wot);

  dim3 gg(EMB / 64, MTOT / 64);  // (16, 128)
  gemm_bt_wmma<true><<<gg, 256, 0, stream>>>(xb, Wqt, qraw, MTOT, EMB, EMB);
  gemm_bt_wmma<true><<<gg, 256, 0, stream>>>(xb, Wkt, kraw, MTOT, EMB, EMB);
  gemm_bt_wmma<true><<<gg, 256, 0, stream>>>(xb, Wvt, vraw, MTOT, EMB, EMB);

  rope_reshape_qk<<<(BATCH * NH * SEQ * 32) / 256, 256, 0, stream>>>(qraw, kraw, Qr, Kr);
  transpose_v<<<(BATCH * NH * SEQ * HDIM) / 256, 256, 0, stream>>>(vraw, Vt);

  flash_attn<<<dim3(SEQ / 64, BATCH * NH), 128, 0, stream>>>(Qr, Kr, Vt, AO);

  gemm_bt_wmma<false><<<gg, 256, 0, stream>>>(AO, Wot, d_out, MTOT, EMB, EMB);
}
